// GroupedWeight_MLP_84799834293039
// MI455X (gfx1250) — matrix operand order repaired and checked by reading, not compiled, not run on hardware
//
#include <hip/hip_runtime.h>
#include <hip/hip_bf16.h>

typedef __attribute__((ext_vector_type(16))) __bf16 v16bf;
typedef __attribute__((ext_vector_type(8)))  __bf16 v8bf;
typedef __attribute__((ext_vector_type(4)))  __bf16 v4bf;
typedef __attribute__((ext_vector_type(8)))  float  v8f;
typedef __attribute__((ext_vector_type(4)))  float  v4f;   // 16B-aligned

#define B_ROWS   8192
#define K_DIM    512     // in_g * in_mem
#define N_DIM    512     // out_g * out_mem
#define NEMB     512
#define DIMQ     64      // in_mem * out_mem
#define NROWS    4096    // in_g * out_g

// ---------------------------------------------------------------------------
// Kernel 1: per-code squared norms of the VQ codebook. embed is [DIMQ][NEMB].
// ---------------------------------------------------------------------------
__global__ void embed_norms_kernel(const float* __restrict__ embed,
                                   float* __restrict__ norms) {
  int e = blockIdx.x * blockDim.x + threadIdx.x;   // 0..511
  float s = 0.f;
#pragma unroll 8
  for (int d = 0; d < DIMQ; ++d) {
    float v = embed[d * NEMB + e];
    s = fmaf(v, v, s);
  }
  norms[e] = s;
}

// ---------------------------------------------------------------------------
// Kernel 2: VQ quantize (exact fp32, argmin is precision-critical), diff
// partials, and pack quantized weights as bf16 W^T [N_DIM][K_DIM] for GEMM.
// One wave (32 lanes) per weight row; 8 rows per 256-thread block.
// ---------------------------------------------------------------------------
__global__ __launch_bounds__(256) void quantize_pack_kernel(
    const float* __restrict__ weight, const float* __restrict__ embed,
    const float* __restrict__ norms, __bf16* __restrict__ wt,
    float* __restrict__ partials) {
  __shared__ float wrow[8][DIMQ];
  __shared__ float sdiff;
  const int tid  = threadIdx.x;
  const int lane = tid & 31;
  const int wid  = tid >> 5;
  const int r    = blockIdx.x * 8 + wid;   // w_flat row, 0..4095

  if (tid == 0) sdiff = 0.f;

  // Each lane owns w[2*lane], w[2*lane+1] of its row.
  float2 w2 = *(const float2*)(weight + (size_t)r * DIMQ + lane * 2);
  wrow[wid][2 * lane]     = w2.x;
  wrow[wid][2 * lane + 1] = w2.y;
  __syncthreads();

  // Each lane scores codes e = lane + 32*j, j = 0..15 (coalesced embed reads).
  float acc[16];
#pragma unroll
  for (int j = 0; j < 16; ++j) acc[j] = 0.f;

#pragma unroll 4
  for (int d = 0; d < DIMQ; ++d) {
    float wd = wrow[wid][d];
    const float* erow = embed + (size_t)d * NEMB + lane;
#pragma unroll
    for (int j = 0; j < 16; ++j)
      acc[j] = fmaf(wd, erow[32 * j], acc[j]);
  }

  // argmin of ||e||^2 - 2 w.e (||w||^2 constant per row); first-index ties.
  float best = 3.4e38f;
  int   bidx = 0;
#pragma unroll
  for (int j = 0; j < 16; ++j) {
    int e = lane + 32 * j;
    float score = norms[e] - 2.f * acc[j];
    if (score < best) { best = score; bidx = e; }
  }
#pragma unroll
  for (int off = 16; off > 0; off >>= 1) {
    float ob = __shfl_xor(best, off);
    int   oi = __shfl_xor(bidx, off);
    if (ob < best || (ob == best && oi < bidx)) { best = ob; bidx = oi; }
  }
  // After xor-reduction every lane holds the winning (best, bidx).

  // Gather the chosen code for this lane's two dims; diff partial.
  float q0 = embed[(size_t)(2 * lane)     * NEMB + bidx];
  float q1 = embed[(size_t)(2 * lane + 1) * NEMB + bidx];
  float d0 = q0 - w2.x, d1 = q1 - w2.y;
  float ds = fmaf(d0, d0, d1 * d1);
#pragma unroll
  for (int off = 16; off > 0; off >>= 1) ds += __shfl_xor(ds, off);
  if (lane == 0) atomicAdd(&sdiff, ds);   // ds_add_f32 on LDS

  // Pack: r = g*64+og; flat dim dd = i*8+om -> W^T[o=og*8+om][k=g*8+i].
  const int g = r >> 6, og = r & 63;
  {
    int dd = 2 * lane;
    wt[(size_t)(og * 8 + (dd & 7)) * K_DIM + g * 8 + (dd >> 3)] = (__bf16)q0;
    dd = 2 * lane + 1;
    wt[(size_t)(og * 8 + (dd & 7)) * K_DIM + g * 8 + (dd >> 3)] = (__bf16)q1;
  }
  __syncthreads();
  if (tid == 0) partials[blockIdx.x] = sdiff;
}

// ---------------------------------------------------------------------------
// Kernel 3: res = x @ W via v_wmma_f32_16x16x32_bf16 (fp32 accumulate).
// 128x128 block tile, 8 waves: wave tile 32(m) x 64(n) = 2x4 WMMA tiles.
// Double-buffered LDS; A staged fp32->bf16 through VGPRs (b128 loads);
// B staged with CDNA5 async global->LDS copies (ASYNCcnt) from the
// pre-packed bf16 W^T [N][K].
// ---------------------------------------------------------------------------
#define TM 128
#define TN 128
#define TK 32
#define LDA 40
#define LDB 40
#define NITER (K_DIM / TK)

__device__ __forceinline__ void async_g2l_b128(unsigned lds_off, unsigned voff,
                                               unsigned long long sbase) {
  asm volatile("global_load_async_to_lds_b128 %0, %1, %2"
               :: "v"(lds_off), "v"(voff), "s"(sbase) : "memory");
}
__device__ __forceinline__ void wait_asynccnt0() {
  asm volatile("s_wait_asynccnt 0" ::: "memory");
}

__device__ __forceinline__ void load_a_regs(const float* __restrict__ x,
                                            int bm, int k0, int tid,
                                            v4f* areg) {
#pragma unroll
  for (int i = 0; i < 4; ++i) {
    int c = tid + i * 256;           // 0..1023
    int row = c >> 3;
    int kc  = (c & 7) << 2;
    areg[i] = *(const v4f*)(x + (size_t)(bm * TM + row) * K_DIM + k0 + kc);
  }
}

__device__ __forceinline__ void store_a_lds(__bf16* __restrict__ abuf, int tid,
                                            const v4f* areg) {
#pragma unroll
  for (int i = 0; i < 4; ++i) {
    int c = tid + i * 256;
    int row = c >> 3;
    int kc  = (c & 7) << 2;
    v4bf b4 = { (__bf16)areg[i][0], (__bf16)areg[i][1],
                (__bf16)areg[i][2], (__bf16)areg[i][3] };
    *(v4bf*)(abuf + row * LDA + kc) = b4;
  }
}

__device__ __forceinline__ void stage_b_async(__bf16* __restrict__ bbuf,
                                              const __bf16* __restrict__ wt,
                                              int bn, int k0, int tid) {
  unsigned long long base = (unsigned long long)(uintptr_t)wt;
#pragma unroll
  for (int i = 0; i < 2; ++i) {
    int c = tid + i * 256;           // 0..511
    int o  = c >> 2;
    int k8 = (c & 3) << 3;
    unsigned lds_off = (unsigned)(uintptr_t)(bbuf + o * LDB + k8);
    unsigned voff    = (unsigned)(((bn * TN + o) * K_DIM + k0 + k8) *
                                  (int)sizeof(__bf16));
    async_g2l_b128(lds_off, voff, base);
  }
}

__global__ __launch_bounds__(256) void gemm_wmma_kernel(
    const float* __restrict__ x, const __bf16* __restrict__ wt,
    float* __restrict__ out) {
  __shared__ __bf16 a_lds[2][TM * LDA];   // [buf][row][k] bf16
  __shared__ __bf16 b_lds[2][TN * LDB];   // [buf][col][k] bf16

  const int tid  = threadIdx.x;
  const int lane = tid & 31;
  const int wid  = tid >> 5;
  const int wm   = wid & 3;    // 0..3 -> 32-row slab
  const int wn   = wid >> 2;   // 0..1 -> 64-col slab
  const int bm   = blockIdx.x; // 0..63
  const int bn   = blockIdx.y; // 0..3

  v8f acc[2][4];
#pragma unroll
  for (int mt = 0; mt < 2; ++mt)
#pragma unroll
    for (int nt = 0; nt < 4; ++nt)
      acc[mt][nt] = (v8f){0.f, 0.f, 0.f, 0.f, 0.f, 0.f, 0.f, 0.f};

  const int rsel = lane & 15;
  const int hsel = lane >> 4;

  // ---- prologue: stage buffer 0 with k0 = 0 ----
  {
    v4f areg[4];
    load_a_regs(x, bm, 0, tid, areg);
    stage_b_async(&b_lds[0][0], wt, bn, 0, tid);
    store_a_lds(&a_lds[0][0], tid, areg);
    wait_asynccnt0();
    __syncthreads();
  }

  for (int it = 0; it < NITER; ++it) {
    const int cur = it & 1;
    const int nxt = cur ^ 1;
    const bool pre = (it + 1 < NITER);

    // Issue next tile's loads first so they fly under the WMMAs.
    v4f areg[4];
    if (pre) {
      load_a_regs(x, bm, (it + 1) * TK, tid, areg);
      stage_b_async(&b_lds[nxt][0], wt, bn, (it + 1) * TK, tid);
    }

    // A frag, 16-bit 16x32 layout: lane<16 -> K0..7 & K16..23 of row rsel;
    // lane>=16 -> K8..15 & K24..31.
    v16bf afrag[2], bfrag[4];
#pragma unroll
    for (int mt = 0; mt < 2; ++mt) {
      const __bf16* p = &a_lds[cur][0] + (wm * 32 + mt * 16 + rsel) * LDA + hsel * 8;
      v8bf lo = *(const v8bf*)p;
      v8bf hi = *(const v8bf*)(p + 16);
      afrag[mt] = __builtin_shufflevector(lo, hi,
          0,1,2,3,4,5,6,7,8,9,10,11,12,13,14,15);
    }
    // B frag, 32x16: lanes 0-15 hold K0..7 & K16..23 of col rsel, lanes 16-31 K8..15 & K24..31.
#pragma unroll
    for (int nt = 0; nt < 4; ++nt) {
      const __bf16* p = &b_lds[cur][0] + (wn * 64 + nt * 16 + rsel) * LDB + hsel * 8;
      v8bf lo = *(const v8bf*)p;
      v8bf hi = *(const v8bf*)(p + 16);
      bfrag[nt] = __builtin_shufflevector(lo, hi,
          0,1,2,3,4,5,6,7,8,9,10,11,12,13,14,15);
    }

#pragma unroll
    for (int mt = 0; mt < 2; ++mt)
#pragma unroll
      for (int nt = 0; nt < 4; ++nt)
        acc[mt][nt] = __builtin_amdgcn_wmma_f32_16x16x32_bf16(
            false, afrag[mt], false, bfrag[nt],
            (short)0, acc[mt][nt], false, false);

    if (pre) {
      store_a_lds(&a_lds[nxt][0], tid, areg);   // waits loadcnt internally
      wait_asynccnt0();                          // B async copies done
      __syncthreads();                           // publish buffer nxt
    }
  }

  // Store: D layout — VGPR j: row j (lanes 0-15) / row j+8 (lanes 16-31),
  // col = lane & 15.
#pragma unroll
  for (int mt = 0; mt < 2; ++mt) {
#pragma unroll
    for (int nt = 0; nt < 4; ++nt) {
      int col  = bn * TN + wn * 64 + nt * 16 + rsel;
      int row0 = bm * TM + wm * 32 + mt * 16 + hsel * 8;
#pragma unroll
      for (int j = 0; j < 8; ++j)
        out[(size_t)(row0 + j) * N_DIM + col] = acc[mt][nt][j];
    }
  }
}

// ---------------------------------------------------------------------------
// Kernel 4: reduce 512 diff partials, write mean to d_out[B*N].
// ---------------------------------------------------------------------------
__global__ void finalize_diff_kernel(const float* __restrict__ partials,
                                     float* __restrict__ diff_out) {
  __shared__ float red[256];
  int tid = threadIdx.x;
  float s = partials[tid] + partials[tid + 256];
  red[tid] = s;
  __syncthreads();
  for (int off = 128; off > 0; off >>= 1) {
    if (tid < off) red[tid] += red[tid + off];
    __syncthreads();
  }
  if (tid == 0) *diff_out = red[0] * (1.f / (float)(NROWS * DIMQ));
}

// ---------------------------------------------------------------------------
extern "C" void kernel_launch(void* const* d_in, const int* in_sizes, int n_in,
                              void* d_out, int out_size, void* d_ws, size_t ws_size,
                              hipStream_t stream) {
  (void)in_sizes; (void)n_in; (void)out_size; (void)ws_size;
  const float* x      = (const float*)d_in[0];
  const float* weight = (const float*)d_in[1];
  const float* embed  = (const float*)d_in[2];
  // d_in[3] = use_qw (always 1 in this harness)

  float* out      = (float*)d_out;
  float* diff_out = out + (size_t)B_ROWS * N_DIM;

  // Workspace layout: bf16 W^T [512*512] (512 KB) | norms f32[512] | partials f32[512]
  __bf16* wt       = (__bf16*)d_ws;
  float*  norms    = (float*)((char*)d_ws + (size_t)N_DIM * K_DIM * sizeof(__bf16));
  float*  partials = norms + NEMB;

  embed_norms_kernel<<<dim3(2), dim3(256), 0, stream>>>(embed, norms);
  quantize_pack_kernel<<<dim3(512), dim3(256), 0, stream>>>(weight, embed, norms, wt, partials);
  gemm_wmma_kernel<<<dim3(64, 4), dim3(256), 0, stream>>>(x, wt, out);
  finalize_diff_kernel<<<dim3(1), dim3(256), 0, stream>>>(partials, diff_out);
}
